// FlowDAGNN_14671608283599
// MI455X (gfx1250) — compile-verified, matrix operand order checked by reading, not run on hardware
//
#include <hip/hip_runtime.h>
#include <hip/hip_bf16.h>

// ---------------------------------------------------------------------------
// FlowDAGNN on MI455X (gfx1250, wave32, WMMA).
//  - 8 independent DAGs -> one persistent 512-thread workgroup per graph.
//  - GRU x-side GEMMs hoisted out of the scans into WMMA bf16 GEMMs.
//  - Recurrent h-side GEMV done as M=16-padded v_wmma_f32_16x16x32_bf16
//    (weight-bandwidth bound: padding M is free, weights live as bf16 in L2).
//  - All fragments are 2x b128 loads (ISA 7.12.2 layout: per lane the 16
//    bf16 elements are two contiguous 16B runs). A operands staged in LDS.
//  - All softmax / gate math stays f32 VALU.
// ---------------------------------------------------------------------------

typedef __bf16 bhalf_t;
typedef __attribute__((ext_vector_type(16))) __bf16 v16bf;
typedef __attribute__((ext_vector_type(8)))  float  v8f;

#define HIDC 256          // hidden size
#define GH   768          // 3*HID (GRU gates)
#define MAXN 64           // nodes per graph
#define PREDH 64          // predictor hidden
#define NEGV (-1.0e30f)

union bfrag16 { v16bf v; uint4 q[2]; };

__device__ __forceinline__ float sigmoidf_(float x) { return 1.0f / (1.0f + __expf(-x)); }

__device__ __forceinline__ float dotN_wave(const float* a, const float* b, int n, int lane) {
  float s = 0.0f;
  for (int k = lane; k < n; k += 32) s += a[k] * b[k];
#pragma unroll
  for (int off = 16; off; off >>= 1) s += __shfl_xor(s, off, 32);
  return s;
}

// ---- WMMA fragment builders (ISA 7.12.2 layouts, wave32) -------------------
// 16-bit operand: lane holds row (A: M=lane&15, B: N=lane&15); its 16 bf16
// elements are row[k0+8*half .. +7] and row[k0+16+8*half .. +7]  (half=lane>>4)
// i.e. exactly two 16-byte vector loads.
__device__ __forceinline__ v16bf frag_row(const bhalf_t* row, int k0, int lane) {
  int half = (lane >> 4) & 1;
  bfrag16 u;
  u.q[0] = *(const uint4*)(row + k0 + 8 * half);
  u.q[1] = *(const uint4*)(row + k0 + 16 + 8 * half);
  return u.v;
}

__device__ __forceinline__ v8f wmma_bf16(v16bf a, v16bf b, v8f c) {
  return __builtin_amdgcn_wmma_f32_16x16x32_bf16(false, a, false, b, (short)0, c, false, false);
}

// gi = sA[64,256] @ W^T + bias -> gi[base..base+63][768], WMMA tiled, 16 waves.
// sA is a zero-padded bf16 LDS tile; W is [768][256] bf16 row-major in global.
__device__ __forceinline__ void gemm_gi(const bhalf_t* sA, int kTiles,
                                        const bhalf_t* W, const float* bias,
                                        float* gi, int base, int tid) {
  int wv = tid >> 5, lane = tid & 31;
  for (int t = wv; t < 4 * (GH / 16); t += 16) {
    int mt = t / (GH / 16), nt = t % (GH / 16);
    const bhalf_t* arow = sA + (size_t)(mt * 16 + (lane & 15)) * HIDC;
    const bhalf_t* brow = W + (size_t)(nt * 16 + (lane & 15)) * HIDC;
    v8f acc = {};
    for (int kt = 0; kt < kTiles; ++kt) {
      v16bf a = frag_row(arow, kt * 32, lane);
      v16bf b = frag_row(brow, kt * 32, lane);
      acc = wmma_bf16(a, b, acc);
    }
    int n = nt * 16 + (lane & 15);
    int half = (lane >> 4) & 1;
#pragma unroll
    for (int r = 0; r < 8; ++r) {
      int m = mt * 16 + (half ? r + 8 : r);
      gi[(size_t)(base + m) * GH + n] = acc[r] + bias[n];
    }
  }
}

// gh[768] = msgt_row0[256] @ Whh^T via M=16-padded WMMA; row 0 of D is result.
// msgt is a 16x256 bf16 LDS tile whose rows 1..15 stay zero.
__device__ __forceinline__ void gemv_gh_wmma(const bhalf_t* msgt, const bhalf_t* Whh,
                                             float* gh, int tid) {
  int wv = tid >> 5, lane = tid & 31;
  const bhalf_t* arow = msgt + (size_t)(lane & 15) * HIDC;
  int nt0 = wv * 3;  // 48 N-tiles over 16 waves
  const bhalf_t* b0 = Whh + (size_t)((nt0 + 0) * 16 + (lane & 15)) * HIDC;
  const bhalf_t* b1 = Whh + (size_t)((nt0 + 1) * 16 + (lane & 15)) * HIDC;
  const bhalf_t* b2 = Whh + (size_t)((nt0 + 2) * 16 + (lane & 15)) * HIDC;
  v8f acc0 = {}, acc1 = {}, acc2 = {};
  for (int kt = 0; kt < 8; ++kt) {
    v16bf a = frag_row(arow, kt * 32, lane);
    acc0 = wmma_bf16(a, frag_row(b0, kt * 32, lane), acc0);
    acc1 = wmma_bf16(a, frag_row(b1, kt * 32, lane), acc1);
    acc2 = wmma_bf16(a, frag_row(b2, kt * 32, lane), acc2);
  }
  if (lane < 16) {  // D row 0: VGPR 0, lanes 0..15 (half==0)
    gh[(nt0 + 0) * 16 + lane] = acc0[0];
    gh[(nt0 + 1) * 16 + lane] = acc1[0];
    gh[(nt0 + 2) * 16 + lane] = acc2[0];
  }
}

// ---------------------------------------------------------------------------
__global__ void cvt_wt_kernel(const float* src, bhalf_t* dst, int rows, int scols, int dcols) {
  int idx = blockIdx.x * blockDim.x + threadIdx.x;
  int tot = rows * dcols;
  if (idx >= tot) return;
  int r = idx / dcols, c = idx - r * dcols;
  float v = (c < scols) ? src[(size_t)r * scols + c] : 0.0f;
  dst[idx] = (bhalf_t)v;
}

// ---------------------------------------------------------------------------
__global__ __launch_bounds__(512) void conv_layer_kernel(
    const float* xin, int ic,
    const bhalf_t* gb_wih_b, const float* gb_bih, const bhalf_t* gb_whh_b, const float* gb_bhh,
    const bhalf_t* gf_wih_b, const float* gf_bih, const bhalf_t* gf_whh_b, const float* gf_bhh,
    const float* att_wv, const float* att_bv, const float* att_wp, const float* att_bp,
    const float* fp_ws, const float* fp_bs, const float* fp_wv, const float* fp_bv,
    const int* sources, const int* targets, const int* maxNodesPtr,
    const int* esrc, const int* edst, int nE, int Ntot,
    float* gi, float* hbw, float* hfw) {
  __shared__ __align__(16) bhalf_t s_xa[MAXN * HIDC];   // staged A operand (bf16, zero-pad)
  __shared__ __align__(16) bhalf_t s_msgt[16 * HIDC];   // msg tile, rows 1..15 zero
  __shared__ float s_aw[MAXN * MAXN];        // att_w (bw) then flow (fw)
  __shared__ unsigned long long s_adj[MAXN]; // successor bitmask per node
  __shared__ float s_vec[MAXN];              // hv (bw) then hs (fw)
  __shared__ float s_msg[HIDC];
  __shared__ float s_gh[GH];
  __shared__ float s_t[MAXN];
  __shared__ float s_sc[2];

  const int tid = threadIdx.x, g = blockIdx.x;
  const int lane = tid & 31, wv = tid >> 5;
  const int base = sources[g];
  const int npg = targets[g] - base + 1;
  const int maxN = maxNodesPtr[0];

  // init state
  for (int i = tid; i < MAXN * HIDC; i += 512) {
    hbw[(size_t)base * HIDC + i] = 0.0f;
    hfw[(size_t)base * HIDC + i] = 0.0f;
  }
  for (int i = tid; i < MAXN * MAXN; i += 512) s_aw[i] = 0.0f;
  for (int i = tid; i < 16 * HIDC; i += 512) s_msgt[i] = (bhalf_t)0.0f;
  if (tid < MAXN) s_adj[tid] = 0ull;
  __syncthreads();
  for (int e = tid; e < nE; e += 512) {
    int s = esrc[e];
    if (s >= base && s < base + MAXN) {
      int d = edst[e] - base;
      if (d >= 0 && d < MAXN) atomicOr(&s_adj[s - base], 1ull << d);
    }
  }
  // hv[u] = x_bw[u].att_wv + bv (loop-invariant over the bw scan)
  for (int u = wv; u < MAXN; u += 16) {
    float d = dotN_wave(xin + (size_t)(base + u) * ic, att_wv, ic, lane) + att_bv[0];
    if (lane == 0) s_vec[u] = d;
  }
  // stage x_bw as zero-padded bf16 tile, then gi_bw = x_bw @ gb_wih^T + bih
  for (int i = tid; i < MAXN * HIDC; i += 512) {
    int r = i >> 8, c = i & (HIDC - 1);
    s_xa[i] = (c < ic) ? (bhalf_t)xin[(size_t)(base + r) * ic + c] : (bhalf_t)0.0f;
  }
  __syncthreads();
  gemm_gi(s_xa, (ic + 31) / 32, gb_wih_b, gb_bih, gi, base, tid);
  __threadfence_block();
  __syncthreads();

  // ============================ backward scan ==============================
  for (int i = 0; i < maxN; ++i) {
    if (i >= npg) continue;  // block-uniform
    const int v = (targets[g] - base) - i;
    if (tid == 0) __builtin_prefetch(gi + (size_t)(base + v) * GH, 0, 1);
    if (wv == 0) {  // hp[v] = h_bw[v].att_wp + bp (h row still zero, compute faithfully)
      float d = dotN_wave(hbw + (size_t)(base + v) * HIDC, att_wp, HIDC, lane) + att_bp[0];
      if (lane == 0) s_sc[0] = d;
    }
    __syncthreads();
    // att_w update: column-u softmax has a single masked-in entry (row v).
    //   s != 0 : softmax -> exactly 1 (exp(-1e30-s) underflows to 0)
    //   s == 0 : entry forced to -1e30 -> all-equal column -> uniform 1/Ntot
    if (tid < MAXN) {
      int u = tid;
      if ((s_adj[u] >> v) & 1ull) {
        float s = s_vec[u] + s_sc[0];
        s_aw[v * MAXN + u] += (s != 0.0f) ? 1.0f : (1.0f / (float)Ntot);
      }
    }
    __syncthreads();
    // msg[v] = sum_u att_w[u][v] * h_bw[u]
    if (tid < HIDC) {
      float acc = 0.0f;
      for (int u = 0; u < MAXN; ++u) {
        float c = s_aw[u * MAXN + v];
        if (c != 0.0f) acc += c * hbw[(size_t)(base + u) * HIDC + tid];
      }
      s_msg[tid] = acc;
      s_msgt[tid] = (bhalf_t)acc;  // row 0 of the WMMA A tile
    }
    __syncthreads();
    gemv_gh_wmma(s_msgt, gb_whh_b, s_gh, tid);
    __syncthreads();
    if (tid < HIDC) {  // GRU gates, f32
      const float* gir = gi + (size_t)(base + v) * GH;
      float ir = gir[tid], iz = gir[HIDC + tid], ig = gir[2 * HIDC + tid];
      float hr = s_gh[tid] + gb_bhh[tid];
      float hz = s_gh[HIDC + tid] + gb_bhh[HIDC + tid];
      float hg = s_gh[2 * HIDC + tid] + gb_bhh[2 * HIDC + tid];
      float r = sigmoidf_(ir + hr), z = sigmoidf_(iz + hz);
      float gg = tanhf(ig + r * hg);
      hbw[(size_t)(base + v) * HIDC + tid] += (1.0f - z) * gg + z * s_msg[tid];
    }
    __threadfence_block();
    __syncthreads();
  }

  // ===================== between scans: flow=0, hs, gi_fw ==================
  for (int i = tid; i < MAXN * MAXN; i += 512) s_aw[i] = 0.0f;
  for (int u = wv; u < MAXN; u += 16) {
    float d = dotN_wave(hbw + (size_t)(base + u) * HIDC, fp_ws, HIDC, lane) + fp_bs[0];
    if (lane == 0) s_vec[u] = d;
  }
  __syncthreads();
  // stage h_bw as bf16 tile, then gi_fw = h_bw @ gf_wih^T + bih
  for (int i = tid; i < MAXN * HIDC; i += 512)
    s_xa[i] = (bhalf_t)hbw[(size_t)base * HIDC + i];
  __syncthreads();
  gemm_gi(s_xa, 8, gf_wih_b, gf_bih, gi, base, tid);
  __threadfence_block();
  __syncthreads();

  // ============================ forward scan ===============================
  for (int i = 0; i < maxN; ++i) {
    if (i >= npg) continue;
    const int v = i;
    if (tid == 0) __builtin_prefetch(gi + (size_t)(base + v) * GH, 0, 1);
    if (tid < HIDC) {  // msg
      float acc;
      if (i == 0) {
        acc = hbw[(size_t)(base + v) * HIDC + tid];
      } else {
        acc = 0.0f;
        for (int w = 0; w < MAXN; ++w) {
          float c = s_aw[w * MAXN + v];
          if (c != 0.0f) acc += c * hfw[(size_t)(base + w) * HIDC + tid];
        }
      }
      s_msg[tid] = acc;
      s_msgt[tid] = (bhalf_t)acc;
    }
    __syncthreads();
    gemv_gh_wmma(s_msgt, gf_whh_b, s_gh, tid);
    __syncthreads();
    if (tid < HIDC) {
      const float* gir = gi + (size_t)(base + v) * GH;
      float ir = gir[tid], iz = gir[HIDC + tid], ig = gir[2 * HIDC + tid];
      float hr = s_gh[tid] + gf_bhh[tid];
      float hz = s_gh[HIDC + tid] + gf_bhh[HIDC + tid];
      float hg = s_gh[2 * HIDC + tid] + gf_bhh[2 * HIDC + tid];
      float r = sigmoidf_(ir + hr), z = sigmoidf_(iz + hz);
      float gg = tanhf(ig + r * hg);
      hfw[(size_t)(base + v) * HIDC + tid] += (1.0f - z) * gg + z * s_msg[tid];
    }
    __threadfence_block();
    __syncthreads();
    // flow update: softmax over row v's successors (axis=1)
    for (int u = wv; u < MAXN; u += 16) {  // wave-uniform branches
      if ((s_adj[v] >> u) & 1ull) {
        float d = dotN_wave(hfw + (size_t)(base + u) * HIDC, fp_wv, HIDC, lane) + fp_bv[0];
        float s = s_vec[v] + d;
        if (lane == 0) s_t[u] = (s != 0.0f) ? s : NEGV;
      }
    }
    __syncthreads();
    if (tid == 0 && i < maxN - 1) {
      unsigned long long bits = s_adj[v];
      float m = -3.0e38f;
      int any = 0;
      for (int u = 0; u < MAXN; ++u)
        if ((bits >> u) & 1ull) { any = 1; if (s_t[u] > m) m = s_t[u]; }
      if (any) {
        if (m <= -1.0e29f) {  // all masked-in scores were exact zeros -> uniform over row
          float c = 1.0f / (float)Ntot;
          for (int u = 0; u < MAXN; ++u)
            if ((bits >> u) & 1ull) s_aw[v * MAXN + u] += c;
        } else {
          float den = 0.0f;
          for (int u = 0; u < MAXN; ++u)
            if ((bits >> u) & 1ull) den += __expf(s_t[u] - m);
          for (int u = 0; u < MAXN; ++u)
            if ((bits >> u) & 1ull) s_aw[v * MAXN + u] += __expf(s_t[u] - m) / den;
        }
      }
    }
    __syncthreads();
  }
}

// ---------------------------------------------------------------------------
__global__ __launch_bounds__(PREDH) void head_kernel(
    const float* hbw1, const float* hfw1, const float* hbw2, const float* hfw2,
    const float* w1, const float* b1, const float* w2, const float* b2,
    const int* sources, const int* targets, float* out) {
  __shared__ float s_h1[PREDH];
  int g = blockIdx.x, j = threadIdx.x;
  int s = sources[g], t = targets[g];
  const float* wr = w1 + (size_t)j * (4 * HIDC);
  float acc = b1[j];
  for (int f = 0; f < HIDC; ++f) acc += hbw1[(size_t)s * HIDC + f] * wr[f];
  for (int f = 0; f < HIDC; ++f) acc += hfw1[(size_t)t * HIDC + f] * wr[HIDC + f];
  for (int f = 0; f < HIDC; ++f) acc += hbw2[(size_t)s * HIDC + f] * wr[2 * HIDC + f];
  for (int f = 0; f < HIDC; ++f) acc += hfw2[(size_t)t * HIDC + f] * wr[3 * HIDC + f];
  s_h1[j] = fmaxf(acc, 0.0f) * w2[j];
  __syncthreads();
  if (j == 0) {
    float o = b2[0];
    for (int k = 0; k < PREDH; ++k) o += s_h1[k];
    out[g] = o;
  }
}

// ---------------------------------------------------------------------------
extern "C" void kernel_launch(void* const* d_in, const int* in_sizes, int n_in,
                              void* d_out, int out_size, void* d_ws, size_t ws_size,
                              hipStream_t stream) {
  (void)n_in; (void)out_size; (void)ws_size;
  // Flattening assumption: setup_inputs() dict in insertion order, params
  // flattened depth-first in insertion order:
  //  0:x, 1..16:layer0{att_wv,att_bv,att_wp,att_bp,fp_ws,fp_bs,fp_wv,fp_bv,
  //                    gf_wih,gf_bih,gf_whh,gf_bhh,gb_wih,gb_bih,gb_whh,gb_bhh},
  //  17..32:layer1{...}, 33:fc_w1, 34:fc_b1, 35:fc_w2, 36:fc_b2,
  //  37:edge_index, 38:batch, 39:sources, 40:targets, 41:max_nodes
  const float* x = (const float*)d_in[0];
  auto LP = [&](int l, int k) { return (const float*)d_in[1 + l * 16 + k]; };
  const float* fc_w1 = (const float*)d_in[33];
  const float* fc_b1 = (const float*)d_in[34];
  const float* fc_w2 = (const float*)d_in[35];
  const float* fc_b2 = (const float*)d_in[36];
  const int* edge = (const int*)d_in[37];
  const int nE = in_sizes[37] / 2;
  const int Ntot = in_sizes[38];
  const int* sources = (const int*)d_in[39];
  const int* targets = (const int*)d_in[40];
  const int* maxNodes = (const int*)d_in[41];
  const int B = in_sizes[39];
  const int ic1 = in_sizes[0] / Ntot;

  // workspace carve
  char* ws = (char*)d_ws;
  size_t off = 0;
  auto alloc = [&](size_t b) { off = (off + 255) & ~(size_t)255; char* p = ws + off; off += b; return p; };
  bhalf_t* wb[2][4];  // 0=gb_wih 1=gb_whh 2=gf_wih 3=gf_whh (K zero-padded to 256)
  for (int l = 0; l < 2; ++l)
    for (int m = 0; m < 4; ++m) wb[l][m] = (bhalf_t*)alloc((size_t)GH * HIDC * 2);
  float* gi   = (float*)alloc((size_t)Ntot * GH * 4);
  float* hbw1 = (float*)alloc((size_t)Ntot * HIDC * 4);
  float* hfw1 = (float*)alloc((size_t)Ntot * HIDC * 4);
  float* hbw2 = (float*)alloc((size_t)Ntot * HIDC * 4);
  float* hfw2 = (float*)alloc((size_t)Ntot * HIDC * 4);

  // bf16 weight conversion (K padded to 256)
  const int tot = GH * HIDC, thr = 256, blk = (tot + thr - 1) / thr;
  for (int l = 0; l < 2; ++l) {
    int icl = (l == 0) ? ic1 : HIDC;
    cvt_wt_kernel<<<blk, thr, 0, stream>>>(LP(l, 12), wb[l][0], GH, icl, HIDC);   // gb_wih
    cvt_wt_kernel<<<blk, thr, 0, stream>>>(LP(l, 14), wb[l][1], GH, HIDC, HIDC);  // gb_whh
    cvt_wt_kernel<<<blk, thr, 0, stream>>>(LP(l, 8),  wb[l][2], GH, HIDC, HIDC);  // gf_wih
    cvt_wt_kernel<<<blk, thr, 0, stream>>>(LP(l, 10), wb[l][3], GH, HIDC, HIDC);  // gf_whh
  }

  // layer 1
  conv_layer_kernel<<<B, 512, 0, stream>>>(
      x, ic1,
      wb[0][0], LP(0, 13), wb[0][1], LP(0, 15),
      wb[0][2], LP(0, 9),  wb[0][3], LP(0, 11),
      LP(0, 0), LP(0, 1), LP(0, 2), LP(0, 3),
      LP(0, 4), LP(0, 5), LP(0, 6), LP(0, 7),
      sources, targets, maxNodes, edge, edge + nE, nE, Ntot,
      gi, hbw1, hfw1);
  // layer 2 (input = h_bw of layer 1)
  conv_layer_kernel<<<B, 512, 0, stream>>>(
      hbw1, HIDC,
      wb[1][0], LP(1, 13), wb[1][1], LP(1, 15),
      wb[1][2], LP(1, 9),  wb[1][3], LP(1, 11),
      LP(1, 0), LP(1, 1), LP(1, 2), LP(1, 3),
      LP(1, 4), LP(1, 5), LP(1, 6), LP(1, 7),
      sources, targets, maxNodes, edge, edge + nE, nE, Ntot,
      gi, hbw2, hfw2);
  // predictor head
  head_kernel<<<B, PREDH, 0, stream>>>(hbw1, hfw1, hbw2, hfw2,
                                       fc_w1, fc_b1, fc_w2, fc_b2,
                                       sources, targets, (float*)d_out);
}
